// DFRNNNoise_57853209477581
// MI455X (gfx1250) — compile-verified, matrix operand order checked by reading, not run on hardware
//
#include <hip/hip_runtime.h>
#include <hip/hip_bf16.h>
#include <stddef.h>

// ---------------------------------------------------------------------------
// Types for CDNA5 WMMA
// ---------------------------------------------------------------------------
typedef __attribute__((ext_vector_type(16))) __bf16       v16bf;
typedef __attribute__((ext_vector_type(8)))  float        v8f;
typedef __attribute__((ext_vector_type(4)))  unsigned int u32x4;
typedef __attribute__((ext_vector_type(8)))  unsigned int u32x8;

#define BB   64
#define TT   512
#define DD   64
#define UU   256
#define GG   1024   // 4*U

__device__ __forceinline__ unsigned short f2bf(float f) {
    unsigned u = __float_as_uint(f);
    u += 0x7FFFu + ((u >> 16) & 1u);          // round-to-nearest-even
    return (unsigned short)(u >> 16);
}
__device__ __forceinline__ float bf2f(unsigned short s) {
    return __uint_as_float(((unsigned)s) << 16);
}
__device__ __forceinline__ float sigmoidf(float x) {
    return 1.0f / (1.0f + __expf(-x));
}

// Build a 16x32 bf16 A/B fragment for lane `lane`:
//   lanes 0-15  hold K = [k0 .. k0+7]  and [k0+16 .. k0+23]
//   lanes 16-31 hold K = [k0+8 ..k0+15] and [k0+24 .. k0+31]
// `p` points at (row_of_this_lane, k0); `half` = lane>>4.
__device__ __forceinline__ v16bf load_frag(const unsigned short* p, int half) {
    u32x4 lo = *(const u32x4*)(p + half * 8);
    u32x4 hi = *(const u32x4*)(p + 16 + half * 8);
    u32x8 w  = __builtin_shufflevector(lo, hi, 0, 1, 2, 3, 4, 5, 6, 7);
    return __builtin_bit_cast(v16bf, w);
}

__device__ __forceinline__ v8f wmma_bf16(v16bf a, v16bf b, v8f c) {
    return __builtin_amdgcn_wmma_f32_16x16x32_bf16(
        /*neg_a=*/false, a, /*neg_b=*/false, b,
        /*c_mod=*/(short)0, c, /*reuse_a=*/false, /*reuse_b=*/false);
}

// ---------------------------------------------------------------------------
// Kernel 0: convert x -> bf16 (row-major [B][T][64]); build transposed bf16
// weights Wt [1024][64] and Ut [1024][256]  (gate-column-major => contiguous
// K per gate column, so B-fragments load like A-fragments).
// ---------------------------------------------------------------------------
#define XN (BB * TT * DD)
#define WN (GG * DD)
#define UN (GG * UU)

__global__ void lstm_prep_kernel(const float* __restrict__ x,
                                 const float* __restrict__ W,
                                 const float* __restrict__ U,
                                 unsigned short* __restrict__ xbf,
                                 unsigned short* __restrict__ Wt,
                                 unsigned short* __restrict__ Ut) {
    int stride = gridDim.x * blockDim.x;
    for (int i = blockIdx.x * blockDim.x + threadIdx.x; i < XN + WN + UN; i += stride) {
        if (i < XN) {
            xbf[i] = f2bf(x[i]);
        } else if (i < XN + WN) {
            int j = i - XN;
            int n = j / DD, k = j % DD;                 // Wt[n][k] = W[k][n]
            Wt[n * DD + k] = f2bf(W[(size_t)k * GG + n]);
        } else {
            int j = i - XN - WN;
            int n = j / UU, k = j % UU;                 // Ut[n][k] = U[k][n]
            Ut[n * UU + k] = f2bf(U[(size_t)k * GG + n]);
        }
    }
}

// ---------------------------------------------------------------------------
// Kernel 1: persistent LSTM recurrence. 4 blocks x 1024 threads (32 waves).
// Block bb owns batch rows [16*bb, 16*bb+16). Wave w owns z-column strips
// 2w and 2w+1 (16 columns each), with its W/U B-fragments held in VGPRs.
// ---------------------------------------------------------------------------
__global__ __launch_bounds__(1024) void lstm_recur_kernel(
        const unsigned short* __restrict__ xbf,
        const unsigned short* __restrict__ Wt,
        const unsigned short* __restrict__ Ut,
        const float* __restrict__ bias,
        unsigned short* __restrict__ hs) {
    extern __shared__ char smem[];
    float*          zs  = (float*)smem;                          // [16][1024] f32
    unsigned short* hsh = (unsigned short*)(smem + 65536);       // [16][256] bf16
    float*          bsh = (float*)(smem + 65536 + 8192);         // [1024] f32

    const int tid  = threadIdx.x;
    const int lane = tid & 31;
    const int wave = tid >> 5;
    const int b0   = blockIdx.x * 16;

    // init: h_0 = 0, stage bias
    for (int i = tid; i < 16 * UU; i += 1024) hsh[i] = 0;
    if (tid < GG) bsh[tid] = bias[tid];
    __syncthreads();

    const int m    = lane & 15;   // M row (matmul) / N col (weights) for this lane
    const int half = lane >> 4;

    // ---- preload B-fragments of U (K=256 -> 8 frags) and W (K=64 -> 2 frags)
    v16bf ufr[2][8];
    v16bf wfr[2][2];
#pragma unroll
    for (int s = 0; s < 2; ++s) {
        const int n = (2 * wave + s) * 16 + m;        // absolute z column
        const unsigned short* urow = Ut + (size_t)n * UU;
        const unsigned short* wrow = Wt + (size_t)n * DD;
#pragma unroll
        for (int kk = 0; kk < 8; ++kk) ufr[s][kk] = load_frag(urow + kk * 32, half);
#pragma unroll
        for (int kk = 0; kk < 2; ++kk) wfr[s][kk] = load_frag(wrow + kk * 32, half);
    }

    // per-thread cell state for the elementwise phase
    const int erow = tid >> 6;          // 0..15
    const int eu   = (tid & 63) * 4;    // 0..252 step 4
    float cst[4] = {0.f, 0.f, 0.f, 0.f};

    const int col0 = 2 * wave * 16 + m;
    const int col1 = col0 + 16;

    for (int t = 0; t < TT; ++t) {
        // ---------------- matmul phase: z = [x_t ; h] @ [W ; U] -------------
        const unsigned short* xrow =
            xbf + ((size_t)(b0 + m) * TT + t) * DD;
        v16bf xa0 = load_frag(xrow, half);
        v16bf xa1 = load_frag(xrow + 32, half);
        if (t + 1 < TT) __builtin_prefetch(xrow + DD, 0, 0);   // next timestep

        v8f acc0 = {};
        v8f acc1 = {};
        acc0 = wmma_bf16(xa0, wfr[0][0], acc0);
        acc1 = wmma_bf16(xa0, wfr[1][0], acc1);
        acc0 = wmma_bf16(xa1, wfr[0][1], acc0);
        acc1 = wmma_bf16(xa1, wfr[1][1], acc1);
#pragma unroll
        for (int kk = 0; kk < 8; ++kk) {
            v16bf ha = load_frag(hsh + m * UU + kk * 32, half);
            acc0 = wmma_bf16(ha, ufr[0][kk], acc0);
            acc1 = wmma_bf16(ha, ufr[1][kk], acc1);
        }
        // C/D layout: VGPR r holds M = r + 8*half, N = lane&15
#pragma unroll
        for (int r = 0; r < 8; ++r) {
            zs[(r + 8 * half) * GG + col0] = acc0[r];
            zs[(r + 8 * half) * GG + col1] = acc1[r];
        }
        __syncthreads();

        // ---------------- elementwise gate phase ----------------------------
        const float* zrow = zs + erow * GG;
        unsigned short hp[4];
#pragma unroll
        for (int j = 0; j < 4; ++j) {
            const int u = eu + j;
            float zi = zrow[u]            + bsh[u];
            float zf = zrow[UU + u]       + bsh[UU + u];
            float zg = zrow[2 * UU + u]   + bsh[2 * UU + u];
            float zo = zrow[3 * UU + u]   + bsh[3 * UU + u];
            float cn = sigmoidf(zf) * cst[j] + sigmoidf(zi) * fmaxf(zg, 0.f);
            cst[j]   = cn;
            hp[j]    = f2bf(sigmoidf(zo) * fmaxf(cn, 0.f));
        }
        ushort4 pk = make_ushort4(hp[0], hp[1], hp[2], hp[3]);
        *reinterpret_cast<ushort4*>(hsh + erow * UU + eu) = pk;
        *reinterpret_cast<ushort4*>(
            hs + ((size_t)(b0 + erow) * TT + t) * UU + eu) = pk;
        __syncthreads();
    }
}

// ---------------------------------------------------------------------------
// Kernel 2: out[b,t] = hs[b,t,:] . dense_w + dense_b   (one wave per (b,t))
// ---------------------------------------------------------------------------
__global__ void lstm_head_kernel(const unsigned short* __restrict__ hs,
                                 const float* __restrict__ dw,
                                 const float* __restrict__ db,
                                 float* __restrict__ out) {
    int idx  = blockIdx.x * 8 + (threadIdx.x >> 5);   // == b*T + t
    int lane = threadIdx.x & 31;
    if (idx >= BB * TT) return;
    const unsigned short* hrow = hs + (size_t)idx * UU;
    float acc = 0.f;
#pragma unroll
    for (int j = 0; j < 8; ++j) {
        int u = lane + j * 32;
        acc += bf2f(hrow[u]) * dw[u];
    }
#pragma unroll
    for (int off = 16; off >= 1; off >>= 1) acc += __shfl_xor(acc, off, 32);
    if (lane == 0) out[idx] = acc + db[0];
}

// ---------------------------------------------------------------------------
extern "C" void kernel_launch(void* const* d_in, const int* in_sizes, int n_in,
                              void* d_out, int out_size, void* d_ws, size_t ws_size,
                              hipStream_t stream) {
    const float* x  = (const float*)d_in[0];
    const float* W  = (const float*)d_in[1];
    const float* U  = (const float*)d_in[2];
    const float* b  = (const float*)d_in[3];
    const float* dw = (const float*)d_in[4];
    const float* db = (const float*)d_in[5];
    float* out = (float*)d_out;

    char* ws = (char*)d_ws;
    unsigned short* xbf = (unsigned short*)(ws);                         // 4 MiB
    unsigned short* Wt  = (unsigned short*)(ws + 4194304);               // 128 KiB
    unsigned short* Ut  = (unsigned short*)(ws + 4194304 + 131072);     // 512 KiB
    unsigned short* hs  = (unsigned short*)(ws + 4194304 + 131072 + 524288); // 16 MiB

    lstm_prep_kernel<<<2048, 1024, 0, stream>>>(x, W, U, xbf, Wt, Ut);

    const int smem = 65536 + 8192 + 4096;   // z + h + bias = 76 KiB (dynamic LDS)
    lstm_recur_kernel<<<4, 1024, smem, stream>>>(xbf, Wt, Ut, b, hs);

    lstm_head_kernel<<<4096, 256, 0, stream>>>(hs, dw, db, out);
}
